// STBlock_63660005261771
// MI455X (gfx1250) — matrix operand order repaired and checked by reading, not compiled, not run on hardware
//
#include <hip/hip_runtime.h>
#include <hip/hip_bf16.h>
#include <math.h>

// ---------------------------------------------------------------------------
// MTGNN-style ST block for MI455X (gfx1250), wave32 + WMMA bf16.
// States kept TRANSPOSED [b, c, l, v] (node dim contiguous) so both WMMA
// operands of the dominant graph GEMM are contiguous global_load_b128s:
//   g^T[l,v] = sum_w  y^T[l,w] (A-op, contiguous K)  *  A_norm[v,w] (B-op,
//   contiguous K = one adjacency row per lane).
// Each wave computes a 32(l) x 64(v) strip: 2 A-frags x 4 B-frags -> 8 WMMAs
// per K-step at 12 b128 loads (1.5 loads/WMMA).
// ---------------------------------------------------------------------------

typedef __bf16 bf16;
typedef __attribute__((ext_vector_type(16))) __bf16 v16bf;
typedef __attribute__((ext_vector_type(8)))  __bf16 v8bf;
typedef __attribute__((ext_vector_type(8)))  float  v8f;

#define BATCH  8
#define CH     64
#define NNODE  512
#define TIN    64
#define LOUT   58
#define SLICES (BATCH * CH)          // 512 (b,c) GEMM slices
#define PLANE  (LOUT * NNODE)        // 29696 (l,v) plane, divisible by 512
#define STATE_ELEMS ((size_t)SLICES * PLANE)

// ---------------------------------------------------------------------------
// helpers
// ---------------------------------------------------------------------------
__global__ __launch_bounds__(256) void zero_f32_kernel(float* p, int n) {
  int i = blockIdx.x * 256 + threadIdx.x;
  if (i < n) p[i] = 0.0f;
}

__global__ __launch_bounds__(256) void cvt_bf16_kernel(const float* __restrict__ s,
                                                       bf16* __restrict__ d, int n) {
  int i = blockIdx.x * 256 + threadIdx.x;
  if (i < n) d[i] = (bf16)s[i];
}

// degree^-1/2 of (A+I) for A and A^T
__global__ __launch_bounds__(256) void degree_kernel(const float* __restrict__ adj,
                                                     float* __restrict__ rinv,
                                                     float* __restrict__ cinv) {
  int t = blockIdx.x * 256 + threadIdx.x;
  if (t < NNODE) {
    float s = 1.0f;
    for (int w = 0; w < NNODE; ++w) s += adj[t * NNODE + w];
    rinv[t] = rsqrtf(s);
  } else if (t < 2 * NNODE) {
    int c = t - NNODE;
    float s = 1.0f;
    for (int v = 0; v < NNODE; ++v) s += adj[v * NNODE + c];
    cinv[c] = rsqrtf(s);
  }
}

// An = D^-1/2 (A+I) D^-1/2 ; AnT likewise for A^T, emitted in bf16 for WMMA
__global__ __launch_bounds__(256) void normadj_kernel(const float* __restrict__ adj,
                                                      const float* __restrict__ rinv,
                                                      const float* __restrict__ cinv,
                                                      bf16* __restrict__ An,
                                                      bf16* __restrict__ AnT) {
  int i = blockIdx.x * 256 + threadIdx.x;       // 512*512 threads
  int v = i >> 9, w = i & (NNODE - 1);
  float diag = (v == w) ? 1.0f : 0.0f;
  An[i]  = (bf16)(rinv[v] * (adj[i] + diag) * rinv[w]);
  AnT[i] = (bf16)(cinv[v] * (adj[w * NNODE + v] + diag) * cinv[w]);
}

// ---------------------------------------------------------------------------
// gated inception: h = tanh(inc1(x)) * sigmoid(inc2(x)), stored bf16,
// TRANSPOSED layout h[b, co, l, v]. One block per (b,v); x[b,:,v,:] in LDS.
// ---------------------------------------------------------------------------
__global__ __launch_bounds__(256)
void inception_kernel(const float* __restrict__ x,
                      const float* fw0, const float* fb0, const float* fw1, const float* fb1,
                      const float* fw2, const float* fb2, const float* fw3, const float* fb3,
                      const float* gw0, const float* gb0, const float* gw1, const float* gb1,
                      const float* gw2, const float* gb2, const float* gw3, const float* gb3,
                      bf16* __restrict__ h) {
  __shared__ float xs[CH * TIN];
  const int b = blockIdx.x >> 9;
  const int v = blockIdx.x & (NNODE - 1);
  const int tid = threadIdx.x;

  for (int i = tid; i < CH * TIN; i += 256)
    xs[i] = x[(((size_t)b * CH + (i >> 6)) * NNODE + v) * TIN + (i & 63)];
  __syncthreads();

  const int KS[4] = {2, 3, 6, 7};
  const float* fw[4] = {fw0, fw1, fw2, fw3};
  const float* fb[4] = {fb0, fb1, fb2, fb3};
  const float* gw[4] = {gw0, gw1, gw2, gw3};
  const float* gb[4] = {gb0, gb1, gb2, gb3};

  for (int oi = tid; oi < CH * LOUT; oi += 256) {
    const int co = oi / LOUT;
    const int l  = oi % LOUT;
    const int j  = co >> 4;       // branch (kernel size group)
    const int cg = co & 15;       // channel within group
    const int k  = KS[j];
    const int t0 = l + 7 - k;     // window always ends at l+6
    float f = fb[j][cg];
    float g = gb[j][cg];
    const float* wf = fw[j] + (size_t)cg * CH * k;
    const float* wg = gw[j] + (size_t)cg * CH * k;
    for (int ci = 0; ci < CH; ++ci) {
      const float* xr = xs + ci * TIN + t0;
      for (int t = 0; t < k; ++t) {
        f += xr[t] * wf[ci * k + t];
        g += xr[t] * wg[ci * k + t];
      }
    }
    float hv = tanhf(f) * (1.0f / (1.0f + __expf(-g)));
    // transposed store: [b, co, l, v]
    h[(((size_t)b * CH + co) * LOUT + l) * NNODE + v] = (bf16)hv;
  }
}

// ---------------------------------------------------------------------------
// RK4 graph-propagation stage (transposed):
//   g^T[l, v] = sum_w Yin^T[l, w] * A[v, w]       per (b,c) slice
//   Yout = cg*g + c0*s0 + c1*s1 + c2*s2           (bf16, transposed layout)
// one wave: 32(l) x 64(v) strip, 2 A-frags x 4 B-frags -> 8 WMMAs / K-step.
// All operand loads are contiguous 16B (b128); EXEC stays full everywhere.
// ---------------------------------------------------------------------------
#define MBLK 2
#define NBLK 4
__global__ __launch_bounds__(256)
void gemm_stage_kernel(const bf16* __restrict__ A,     // [512,512] bf16 row-major
                       const bf16* __restrict__ Yin,
                       const bf16* __restrict__ s0, float c0,
                       const bf16* __restrict__ s1, float c1,
                       const bf16* __restrict__ s2, float c2,
                       float cg,
                       bf16* __restrict__ Yout) {
  const int wid  = blockIdx.x * 8 + (threadIdx.x >> 5);
  const int lane = threadIdx.x & 31;
  const int vg    = wid & 7;               // 8 groups of 4 v-tiles = 512 cols
  const int lt    = (wid >> 3) & 1;        // 2 row-tile pairs cover 58 (+mask)
  const int slice = wid >> 4;              // 512 (b,c) slices
  const int v0 = vg * (16 * NBLK);
  const int l0 = lt * (16 * MBLK);
  const int hs = lane >> 4;                // half-wave select
  const int lr = lane & 15;

  const int kbA = hs * 8;    // ISA 16-bit A layout: lanes16-31 start at K=8
  const int kbB = hs * 8;    // same layout as A: lanes16-31 start at K=8

  const bf16* __restrict__ ap[MBLK];
#pragma unroll
  for (int m = 0; m < MBLK; ++m) {
    const int lrow = l0 + m * 16 + lr;
    const int lclamp = (lrow < LOUT) ? lrow : (LOUT - 1);  // clamp, EXEC full
    ap[m] = Yin + ((size_t)slice * LOUT + lclamp) * NNODE + kbA;
  }
  const bf16* __restrict__ bp0 = A + (size_t)(v0 + lr) * NNODE + kbB;

  v8f acc[MBLK][NBLK] = {};
  for (int kc = 0; kc < 16; ++kc) {
    const int w0 = kc * 32;
    // A fragments (state rows): two contiguous 16B runs each
    v16bf af[MBLK];
#pragma unroll
    for (int m = 0; m < MBLK; ++m) {
      const v8bf alo = *(const v8bf*)(ap[m] + w0);
      const v8bf ahi = *(const v8bf*)(ap[m] + w0 + 16);
#pragma unroll
      for (int i = 0; i < 8; ++i) { af[m][i] = alo[i]; af[m][i + 8] = ahi[i]; }
    }
#pragma unroll
    for (int blk = 0; blk < NBLK; ++blk) {
      // B fragment (adjacency row per lane): two contiguous 16B runs
      const bf16* bp = bp0 + (size_t)blk * 16 * NNODE + w0;
      const v8bf blo = *(const v8bf*)(bp);
      const v8bf bhi = *(const v8bf*)(bp + 16);
      v16bf bfr;
#pragma unroll
      for (int i = 0; i < 8; ++i) { bfr[i] = blo[i]; bfr[i + 8] = bhi[i]; }
#pragma unroll
      for (int m = 0; m < MBLK; ++m)
        acc[m][blk] = __builtin_amdgcn_wmma_f32_16x16x32_bf16(
            false, af[m], false, bfr, (short)0, acc[m][blk], false, false);
    }
  }

  // epilogue: rows are l (masked at 58), cols are v (contiguous stores)
#pragma unroll
  for (int m = 0; m < MBLK; ++m) {
#pragma unroll
    for (int r = 0; r < 8; ++r) {
      const int l = l0 + m * 16 + r + hs * 8;  // C/D layout: VGPR r -> r / r+8
      if (l < LOUT) {
#pragma unroll
        for (int blk = 0; blk < NBLK; ++blk) {
          const size_t idx = ((size_t)slice * LOUT + l) * NNODE + v0 + blk * 16 + lr;
          float o = cg * acc[m][blk][r];
          if (s0) o += c0 * (float)s0[idx];
          if (s1) o += c1 * (float)s1[idx];
          if (s2) o += c2 * (float)s2[idx];
          Yout[idx] = (bf16)o;
        }
      }
    }
  }
}

// ---------------------------------------------------------------------------
// 5C->C mixing GEMM: out[b,co,v,l] += sum_k W[co,k]*S[k/64][b,k%64,(l,v)] + b
// K=320 (5 states x 64 ch), N = (l,v) plane (29696 = 1856 full tiles of 16)
// accumulates straight into the front-padded f32 output (both branches add)
// ---------------------------------------------------------------------------
__global__ __launch_bounds__(256)
void mix_kernel(const bf16* __restrict__ W,      // [64,320] bf16
                const float* __restrict__ bias,  // [64] f32
                const bf16* __restrict__ S0, const bf16* __restrict__ S1,
                const bf16* __restrict__ S2, const bf16* __restrict__ S3,
                const bf16* __restrict__ S4,
                float* __restrict__ out) {
  const int wid  = blockIdx.x * 8 + (threadIdx.x >> 5);
  const int lane = threadIdx.x & 31;
  const int nt = wid % 1856;
  const int ct = (wid / 1856) & 3;
  const int b  = wid / (1856 * 4);
  const int co0 = ct * 16;
  const int hs  = lane >> 4;
  const int lr  = lane & 15;
  const int m   = nt * 16 + lr;            // contiguous (l,v) column
  const int arow = co0 + lr;
  const int kbA  = hs * 16;
  const int kbB  = hs * 16;

  v8f acc = {};
#pragma unroll
  for (int s = 0; s < 5; ++s) {
    const bf16* __restrict__ Sp = (s == 0) ? S0 : (s == 1) ? S1 : (s == 2) ? S2
                                  : (s == 3) ? S3 : S4;
    const bf16* __restrict__ sb = Sp + (size_t)b * CH * PLANE + m;
#pragma unroll
    for (int half = 0; half < 2; ++half) {
      const int k0 = s * 64 + half * 32;   // K chunk base
      const int c0 = half * 32 + kbB;      // channel base within this state
      const v8bf alo = *(const v8bf*)(W + (size_t)arow * 320 + k0 + kbA);
      const v8bf ahi = *(const v8bf*)(W + (size_t)arow * 320 + k0 + kbA + 8);
      v16bf af;
#pragma unroll
      for (int i = 0; i < 8; ++i) { af[i] = alo[i]; af[i + 8] = ahi[i]; }
      v16bf bfr;
#pragma unroll
      for (int i = 0; i < 16; ++i) bfr[i] = sb[(size_t)(c0 + i) * PLANE];
      acc = __builtin_amdgcn_wmma_f32_16x16x32_bf16(false, af, false, bfr,
                                                    (short)0, acc, false, false);
    }
  }

  const int l = m >> 9;                    // plane is l-major, v contiguous
  const int v = m & (NNODE - 1);
#pragma unroll
  for (int r = 0; r < 8; ++r) {
    const int co = co0 + r + hs * 8;
    const size_t oidx = (((size_t)b * CH + co) * NNODE + v) * TIN + 6 + l; // pad=6
    out[oidx] += acc[r] + bias[co];
  }
}

// ---------------------------------------------------------------------------
// launcher
// ---------------------------------------------------------------------------
extern "C" void kernel_launch(void* const* d_in, const int* in_sizes, int n_in,
                              void* d_out, int out_size, void* d_ws, size_t ws_size,
                              hipStream_t stream) {
  const float* x   = (const float*)d_in[0];
  const float* adj = (const float*)d_in[1];
  const float* iw[16];
  for (int i = 0; i < 16; ++i) iw[i] = (const float*)d_in[2 + i];  // i1w0..i2b3
  const float* m1w = (const float*)d_in[18];
  const float* m1b = (const float*)d_in[19];
  const float* m2w = (const float*)d_in[20];
  const float* m2b = (const float*)d_in[21];
  float* out = (float*)d_out;

  // workspace carve-up (~153 MB): adj matrices, bf16 weights, 5 bf16 states
  size_t off = 0;
  auto carve = [&](size_t bytes) {
    void* p = (char*)d_ws + off;
    off += (bytes + 255) & ~(size_t)255;
    return p;
  };
  bf16*  An   = (bf16*)carve((size_t)NNODE * NNODE * sizeof(bf16));
  bf16*  AnT  = (bf16*)carve((size_t)NNODE * NNODE * sizeof(bf16));
  float* rinv = (float*)carve(NNODE * sizeof(float));
  float* cinv = (float*)carve(NNODE * sizeof(float));
  bf16*  Wm1  = (bf16*)carve((size_t)CH * 5 * CH * sizeof(bf16));
  bf16*  Wm2  = (bf16*)carve((size_t)CH * 5 * CH * sizeof(bf16));
  const size_t SB = STATE_ELEMS * sizeof(bf16);
  bf16* h  = (bf16*)carve(SB);
  bf16* y2 = (bf16*)carve(SB);
  bf16* y3 = (bf16*)carve(SB);
  bf16* y4 = (bf16*)carve(SB);
  bf16* yf = (bf16*)carve(SB);
  (void)ws_size; (void)in_sizes; (void)n_in;

  // 1) zero padded output (both mix passes accumulate into it)
  zero_f32_kernel<<<(out_size + 255) / 256, 256, 0, stream>>>(out, out_size);

  // 2) normalized adjacencies in bf16
  degree_kernel<<<(2 * NNODE + 255) / 256, 256, 0, stream>>>(adj, rinv, cinv);
  normadj_kernel<<<(NNODE * NNODE) / 256, 256, 0, stream>>>(adj, rinv, cinv, An, AnT);

  // 3) mixing weights to bf16
  cvt_bf16_kernel<<<(CH * 5 * CH + 255) / 256, 256, 0, stream>>>(m1w, Wm1, CH * 5 * CH);
  cvt_bf16_kernel<<<(CH * 5 * CH + 255) / 256, 256, 0, stream>>>(m2w, Wm2, CH * 5 * CH);

  // 4) gated inception -> h (bf16, transposed [b,c,l,v])
  inception_kernel<<<BATCH * NNODE, 256, 0, stream>>>(
      x, iw[0], iw[1], iw[2], iw[3], iw[4], iw[5], iw[6], iw[7],
      iw[8], iw[9], iw[10], iw[11], iw[12], iw[13], iw[14], iw[15], h);

  // 5) two directed branches: 4 WMMA GEMM stages + WMMA mix each
  const int gemmBlocks = (SLICES * 2 * 8) / 8;    // 8192 wave-strips (32x64)
  const int mixBlocks  = (BATCH * 4 * 1856) / 8;  // 59392 wave-tiles
  for (int br = 0; br < 2; ++br) {
    const bf16* A  = br ? AnT : An;
    const bf16* W  = br ? Wm2 : Wm1;
    const float* bias = br ? m2b : m1b;
    // y2 = (2/3)h + (1/3)A.h
    gemm_stage_kernel<<<gemmBlocks, 256, 0, stream>>>(
        A, h, h, 2.0f / 3.0f, nullptr, 0.f, nullptr, 0.f, 1.0f / 3.0f, y2);
    // y3 = 2h - 2y2 + A.y2
    gemm_stage_kernel<<<gemmBlocks, 256, 0, stream>>>(
        A, y2, h, 2.0f, y2, -2.0f, nullptr, 0.f, 1.0f, y3);
    // y4 = 2y2 - 2y3 + A.y3
    gemm_stage_kernel<<<gemmBlocks, 256, 0, stream>>>(
        A, y3, y2, 2.0f, y3, -2.0f, nullptr, 0.f, 1.0f, y4);
    // yf = -h/8 + (3/4)y3 + (1/4)y4 + (1/8)A.y4
    gemm_stage_kernel<<<gemmBlocks, 256, 0, stream>>>(
        A, y4, h, -0.125f, y3, 0.75f, y4, 0.25f, 0.125f, yf);
    // out += W @ [h,y2,y3,y4,yf] + bias
    mix_kernel<<<mixBlocks, 256, 0, stream>>>(W, bias, h, y2, y3, y4, yf, out);
  }
}
